// ARMAConv_71236327571877
// MI455X (gfx1250) — matrix operand order repaired and checked by reading, not compiled, not run on hardware
//
#include <hip/hip_runtime.h>

// ARMAConv on MI455X (gfx1250):
//   2 passes of P = fltr(16384x16384,fp32 -> bf16 in-register) @ Y(16384x64,bf16)
//   via v_wmma_f32_16x16x32_bf16, fused bias/skip + ReLU epilogue.
// HBM-bound: ~2.15 GB total (fltr read twice) ~= 92 us at 23.3 TB/s.

#define NNODES 16384
#define FDIM   64
#define CDIM   32
#define KTOT   NNODES
#define BROWS  64            // rows per workgroup in the big GEMM
#define KB     64            // K per LDS stage (2 WMMA K-steps)
#define ASTRIDE 72           // padded bf16 row stride in LDS

typedef __attribute__((ext_vector_type(16))) __bf16 v16bf;
typedef __attribute__((ext_vector_type(8)))  __bf16 v8bf;
typedef __attribute__((ext_vector_type(8)))  float  v8f;
typedef __attribute__((ext_vector_type(4)))  float  f32x4;

// ---------------------------------------------------------------------------
// Small GEMMs: Yt[c][n] = bf16( (xbar @ w1)[n][c] ), S[n][c] = (x @ w2)[n][c] + b[c]
// c in [0,64): stack k = c/32, within-stack col cc = c%32.
// iter==0: xbar == x (in_dim 64). iter==1: xbar is N x 64 (32 per stack).
// ---------------------------------------------------------------------------
__global__ void __launch_bounds__(256)
arma_prep(const float* __restrict__ x, const float* __restrict__ xbar,
          const float* __restrict__ w1_0, const float* __restrict__ w1_1,
          const float* __restrict__ w2_0, const float* __restrict__ w2_1,
          const float* __restrict__ b_0,  const float* __restrict__ b_1,
          __bf16* __restrict__ Yt, float* __restrict__ S, int iter)
{
    __shared__ float xs[4][FDIM];
    __shared__ float xbs[4][FDIM];
    const int t  = threadIdx.x;
    const int nl = t >> 6;            // 0..3 local node
    const int c  = t & 63;            // 0..63 output column (both stacks)
    const int n  = blockIdx.x * 4 + nl;

    xs[nl][c] = x[n * FDIM + c];
    if (iter == 1) xbs[nl][c] = xbar[n * FDIM + c];
    __syncthreads();

    const int k  = c >> 5;
    const int cc = c & 31;
    const float* w1 = k ? w1_1 : w1_0;
    const float* w2 = k ? w2_1 : w2_0;
    const float* bb = k ? b_1  : b_0;

    float y = 0.f, s = 0.f;
    if (iter == 0) {
        #pragma unroll
        for (int f = 0; f < FDIM; ++f) {
            const float xv = xs[nl][f];
            y += xv * w1[f * CDIM + cc];
            s += xv * w2[f * CDIM + cc];
        }
    } else {
        #pragma unroll
        for (int j = 0; j < CDIM; ++j)
            y += xbs[nl][k * CDIM + j] * w1[j * CDIM + cc];
        #pragma unroll
        for (int f = 0; f < FDIM; ++f)
            s += xs[nl][f] * w2[f * CDIM + cc];
    }
    Yt[c * KTOT + n]       = (__bf16)y;      // transposed: B-fragments are contiguous in K
    S[n * (2 * CDIM) + c]  = s + bb[cc];
}

// ---------------------------------------------------------------------------
// Big GEMM: Out[n][c] = relu( (fltr @ Y)[n][c] + S[n][c] ), Y given as Yt (64 x K bf16).
// WG: 8 waves, 64 rows x 64 cols. Wave w: row group (w>>1)*16, col group (w&1)*32,
// i.e. two 16x16 f32 accumulators per wave.
// ---------------------------------------------------------------------------
__global__ void __launch_bounds__(256)
arma_spmm(const float* __restrict__ fltr, const __bf16* __restrict__ Yt,
          const float* __restrict__ S, float* __restrict__ Out)
{
    __shared__ __align__(16) __bf16 As[BROWS * ASTRIDE];

    const int t    = threadIdx.x;
    const int lane = t & 31;
    const int wv   = t >> 5;
    const int rg   = wv >> 1;                 // 0..3
    const int cg   = wv & 1;                  // 0..1
    const int blockRow = blockIdx.x * BROWS;

    // cooperative staging: thread -> (row sr, 16 consecutive k at sk)
    const int sr = t >> 2;                    // 0..63
    const int sk = (t & 3) << 4;              // 0,16,32,48
    const float* gsrc = fltr + (size_t)(blockRow + sr) * (size_t)KTOT + sk;

    // A fragment addressing (this code's contiguous k = 16h + s order, matching B):
    // lane m=lane&15 holds row m; lanes>=16 shift K by 16; per-lane K chunk hi..hi+15
    const int m  = lane & 15;
    const int hi = (lane >> 4) << 4;          // 0 or 16
    const __bf16* a0base = &As[(rg * 16 + m) * ASTRIDE + hi];

    // B fragment addressing (32x16 B layout): lane holds col m, K chunk (lane>>4)*16 .. +15
    const int khalf = (lane >> 4) << 4;       // 0 or 16
    const __bf16* bbase0 = Yt + (size_t)(cg * 32 + m)      * (size_t)KTOT + khalf;
    const __bf16* bbase1 = Yt + (size_t)(cg * 32 + 16 + m) * (size_t)KTOT + khalf;

    v8f acc0 = {};
    v8f acc1 = {};

    // prologue: load first fltr tile (streamed once -> nontemporal)
    f32x4 r0 = __builtin_nontemporal_load((const f32x4*)(gsrc + 0));
    f32x4 r1 = __builtin_nontemporal_load((const f32x4*)(gsrc + 4));
    f32x4 r2 = __builtin_nontemporal_load((const f32x4*)(gsrc + 8));
    f32x4 r3 = __builtin_nontemporal_load((const f32x4*)(gsrc + 12));

    for (int k0 = 0; k0 < KTOT; k0 += KB) {
        __syncthreads();   // previous tile fully consumed
        // fp32 -> bf16 in-register, store 16 values as two 16B chunks
        v8bf c0, c1;
        c0[0]=(__bf16)r0[0]; c0[1]=(__bf16)r0[1]; c0[2]=(__bf16)r0[2]; c0[3]=(__bf16)r0[3];
        c0[4]=(__bf16)r1[0]; c0[5]=(__bf16)r1[1]; c0[6]=(__bf16)r1[2]; c0[7]=(__bf16)r1[3];
        c1[0]=(__bf16)r2[0]; c1[1]=(__bf16)r2[1]; c1[2]=(__bf16)r2[2]; c1[3]=(__bf16)r2[3];
        c1[4]=(__bf16)r3[0]; c1[5]=(__bf16)r3[1]; c1[6]=(__bf16)r3[2]; c1[7]=(__bf16)r3[3];
        *(v8bf*)(&As[sr * ASTRIDE + sk])     = c0;
        *(v8bf*)(&As[sr * ASTRIDE + sk + 8]) = c1;
        __syncthreads();   // tile visible

        // software pipeline: issue next tile's global loads before the WMMAs
        if (k0 + KB < KTOT) {
            const float* g2 = gsrc + (k0 + KB);
            r0 = __builtin_nontemporal_load((const f32x4*)(g2 + 0));
            r1 = __builtin_nontemporal_load((const f32x4*)(g2 + 4));
            r2 = __builtin_nontemporal_load((const f32x4*)(g2 + 8));
            r3 = __builtin_nontemporal_load((const f32x4*)(g2 + 12));
        }

        #pragma unroll
        for (int kl = 0; kl < KB; kl += 32) {
            const v8bf alo = *(const v8bf*)(a0base + kl);
            const v8bf ahi = *(const v8bf*)(a0base + kl + 8);
            const v16bf afrag = __builtin_shufflevector(
                alo, ahi, 0, 1, 2, 3, 4, 5, 6, 7, 8, 9, 10, 11, 12, 13, 14, 15);
            const v16bf bf0 = *(const v16bf*)(bbase0 + k0 + kl);
            const v16bf bf1 = *(const v16bf*)(bbase1 + k0 + kl);
            acc0 = __builtin_amdgcn_wmma_f32_16x16x32_bf16(
                false, afrag, false, bf0, (short)0, acc0, false, false);
            acc1 = __builtin_amdgcn_wmma_f32_16x16x32_bf16(
                false, afrag, false, bf1, (short)0, acc1, false, false);
        }
    }

    // epilogue: C/D 16x16 f32 layout -> row = base + v + (lane>=16 ? 8 : 0), col = lane&15
    const int row0 = blockRow + rg * 16 + ((lane >> 4) << 3);
    const int col0 = cg * 32 + m;
    #pragma unroll
    for (int v = 0; v < 8; ++v) {
        const int idx0 = (row0 + v) * (2 * CDIM) + col0;
        Out[idx0]      = fmaxf(acc0[v] + S[idx0], 0.f);
        const int idx1 = idx0 + 16;
        Out[idx1]      = fmaxf(acc1[v] + S[idx1], 0.f);
    }
}

// mean over the two stacks
__global__ void __launch_bounds__(256)
arma_final(const float* __restrict__ xbar2, float* __restrict__ out)
{
    const int id = blockIdx.x * 256 + threadIdx.x;
    const int n = id >> 5, c = id & 31;
    out[id] = 0.5f * (xbar2[n * (2 * CDIM) + c] + xbar2[n * (2 * CDIM) + CDIM + c]);
}

extern "C" void kernel_launch(void* const* d_in, const int* in_sizes, int n_in,
                              void* d_out, int out_size, void* d_ws, size_t ws_size,
                              hipStream_t stream)
{
    const float* x       = (const float*)d_in[0];
    const float* fltr    = (const float*)d_in[1];
    const float* w1_k0i0 = (const float*)d_in[2];
    const float* w2_k0i0 = (const float*)d_in[3];
    const float* b_k0i0  = (const float*)d_in[4];
    const float* w1_k0i1 = (const float*)d_in[5];
    const float* w2_k0i1 = (const float*)d_in[6];
    const float* b_k0i1  = (const float*)d_in[7];
    const float* w1_k1i0 = (const float*)d_in[8];
    const float* w2_k1i0 = (const float*)d_in[9];
    const float* b_k1i0  = (const float*)d_in[10];
    const float* w1_k1i1 = (const float*)d_in[11];
    const float* w2_k1i1 = (const float*)d_in[12];
    const float* b_k1i1  = (const float*)d_in[13];

    char* ws = (char*)d_ws;
    __bf16* Yt   = (__bf16*)(ws);                 // 64*16384 bf16 = 2 MiB
    float*  S    = (float*)(ws + (2u  << 20));    // 16384*64 f32  = 4 MiB
    float*  xb1  = (float*)(ws + (6u  << 20));    // 4 MiB
    float*  xb2  = (float*)(ws + (10u << 20));    // 4 MiB

    const dim3 blk(256);
    const dim3 prepGrid(NNODES / 4);
    const dim3 gemmGrid(NNODES / BROWS);
    const dim3 finGrid(NNODES * CDIM / 256);

    // iteration 0 (xbar = x)
    arma_prep<<<prepGrid, blk, 0, stream>>>(x, x, w1_k0i0, w1_k1i0, w2_k0i0, w2_k1i0,
                                            b_k0i0, b_k1i0, Yt, S, 0);
    arma_spmm<<<gemmGrid, blk, 0, stream>>>(fltr, Yt, S, xb1);
    // iteration 1
    arma_prep<<<prepGrid, blk, 0, stream>>>(x, xb1, w1_k0i1, w1_k1i1, w2_k0i1, w2_k1i1,
                                            b_k0i1, b_k1i1, Yt, S, 1);
    arma_spmm<<<gemmGrid, blk, 0, stream>>>(fltr, Yt, S, xb2);
    // mean over stacks
    arma_final<<<finGrid, blk, 0, stream>>>(xb2, (float*)d_out);
}
